// QKVAttention_420906795721
// MI455X (gfx1250) — matrix operand order repaired and checked by reading, not compiled, not run on hardware
//
#include <hip/hip_runtime.h>
#include <hip/hip_bf16.h>
#include <math.h>

// Problem constants (fixed by the reference)
#define N_B 2
#define T_S 2048
#define C_D 2048
#define H_N 16
#define D_H 128

typedef __attribute__((ext_vector_type(16))) _Float16 v16h;
typedef __attribute__((ext_vector_type(8)))  _Float16 v8h;
typedef __attribute__((ext_vector_type(8)))  float    v8f;

// ---------------------------------------------------------------------------
// WMMA fragment loaders (CDNA5 ISA 7.12.2 layouts, wave32)
// A-matrix 16x32 f16: lane holds row m = lane&15; halves are K = kb..kb+7 and
// kb+16..kb+23 with kb = k0 + (lane>=16 ? 8 : 0)  -> two global_load_b128.
// B-matrix 32x16 f16: lane holds col n = lane&15; halves are K = kb..kb+7 and
// kb+16..kb+23 with kb = k0 + (lane>=16 ? 8 : 0)  -> two global_load_b128.
// ---------------------------------------------------------------------------
__device__ __forceinline__ v16h load_a_frag(const _Float16* base, int ld,
                                            int row0, int k0) {
  const int lane = threadIdx.x & 31;
  const int m  = row0 + (lane & 15);
  const int kb = k0 + ((lane & 16) ? 8 : 0);
  const _Float16* p = base + (size_t)m * ld + kb;
  v8h lo = *(const v8h*)(p);
  v8h hi = *(const v8h*)(p + 16);
  v16h f;
#pragma unroll
  for (int i = 0; i < 8; ++i) { f[i] = lo[i]; f[i + 8] = hi[i]; }
  return f;
}

__device__ __forceinline__ v16h load_b_frag(const _Float16* base, int ld,
                                            int col0, int k0) {
  const int lane = threadIdx.x & 31;
  const int n  = col0 + (lane & 15);
  const int kb = k0 + ((lane & 16) ? 8 : 0);
  const _Float16* p = base + (size_t)n * ld + kb;
  v8h lo = *(const v8h*)(p);
  v8h hi = *(const v8h*)(p + 16);
  v16h f;
#pragma unroll
  for (int i = 0; i < 8; ++i) { f[i] = lo[i]; f[i + 8] = hi[i]; }
  return f;
}

__device__ __forceinline__ v8f wmma16(v16h a, v16h b, v8f c) {
  return __builtin_amdgcn_wmma_f32_16x16x32_f16(
      /*neg_a=*/false, a, /*neg_b=*/false, b,
      /*c_mod=*/(short)0, c, /*reuse_a=*/false, /*reuse_b=*/false);
}

// ---------------------------------------------------------------------------
// fp32 -> f16 conversion (one pass; thereafter everything is L2-resident f16)
// ---------------------------------------------------------------------------
__global__ void cvt_f16_kernel(const float* __restrict__ src,
                               _Float16* __restrict__ dst, size_t n) {
  size_t i = (size_t)blockIdx.x * blockDim.x + threadIdx.x;
  const size_t stride = (size_t)gridDim.x * blockDim.x;
  for (; i < n; i += stride) dst[i] = (_Float16)src[i];
}

// transpose-convert: dst[c*rows + r] = src[r*cols + c]   (for w_o)
__global__ void cvt_f16_t_kernel(const float* __restrict__ src,
                                 _Float16* __restrict__ dst, int rows, int cols) {
  size_t i = (size_t)blockIdx.x * blockDim.x + threadIdx.x;
  const size_t n = (size_t)rows * cols;
  const size_t stride = (size_t)gridDim.x * blockDim.x;
  for (; i < n; i += stride) {
    const int r = (int)(i / cols);
    const int c = (int)(i % cols);
    dst[(size_t)c * rows + r] = (_Float16)src[i];
  }
}

// ---------------------------------------------------------------------------
// Generic WMMA GEMM:  Out[m,n] = sum_k A[m,k] * Brow[n,k]
//   (B is supplied as rows of the *output* dimension, contiguous over K,
//    which matches both W[d,c] weights and transposed operands)
// Wave tile: 32(M) x 64(N)  -> 2 A-frags + 4 B-frags -> 8 WMMAs per K-step.
// Optional epilogues: f16 store (normal or transposed) + f32 store.
// Batch: blockIdx.y -> (bn = y/Hdiv applies to A, bh = y%Hdiv applies to B).
// ---------------------------------------------------------------------------
__global__ void __launch_bounds__(256)
wmma_gemm_kernel(const _Float16* __restrict__ A, const _Float16* __restrict__ B,
                 _Float16* out16, float* out32,
                 int M, int N, int K, int lda, int ldb,
                 long aStrideN, long bStrideH, int Hdiv,
                 long o16Stride, int ld16, int trans16,
                 long o32Stride, int ld32) {
  const int batch = blockIdx.y;
  const int bn = batch / Hdiv, bh = batch % Hdiv;
  A += (size_t)bn * aStrideN;
  B += (size_t)bh * bStrideH;
  if (out16) out16 += (size_t)batch * o16Stride;
  if (out32) out32 += (size_t)batch * o32Stride;

  const int wave    = blockIdx.x * (blockDim.x >> 5) + (threadIdx.x >> 5);
  const int ntilesN = N >> 6;
  const int m0 = (wave / ntilesN) << 5;
  const int n0 = (wave % ntilesN) << 6;
  if (m0 >= M) return;

  v8f acc[2][4] = {};

  for (int k0 = 0; k0 < K; k0 += 32) {
    const v16h a0 = load_a_frag(A, lda, m0,      k0);
    const v16h a1 = load_a_frag(A, lda, m0 + 16, k0);
#pragma unroll
    for (int j = 0; j < 4; ++j) {
      const v16h b = load_b_frag(B, ldb, n0 + j * 16, k0);
      acc[0][j] = wmma16(a0, b, acc[0][j]);
      acc[1][j] = wmma16(a1, b, acc[1][j]);
    }
  }

  const int lane = threadIdx.x & 31;
  const int col  = lane & 15;
  const int rb   = (lane & 16) ? 8 : 0;
#pragma unroll
  for (int mi = 0; mi < 2; ++mi)
#pragma unroll
    for (int j = 0; j < 4; ++j)
#pragma unroll
      for (int r = 0; r < 8; ++r) {
        const int m = m0 + mi * 16 + rb + r;
        const int n = n0 + j * 16 + col;
        const float val = acc[mi][j][r];
        if (out32) out32[(size_t)m * ld32 + n] = val;
        if (out16) {
          if (trans16) out16[(size_t)n * ld16 + m] = (_Float16)val;
          else         out16[(size_t)m * ld16 + n] = (_Float16)val;
        }
      }
}

// ---------------------------------------------------------------------------
// Flash-style causal attention, one wave per 16 query rows.
//   Q kept in registers (4 A-frags of 16x32 over D=128).
//   Key blocks of 32: 8 WMMAs QK^T, online softmax (16-lane shfl_xor row
//   reductions, rows of a C-frag span one 16-lane half of the wave),
//   P bounced through per-wave LDS (C-layout -> A-layout), 8 WMMAs P*V.
//   V is pre-transposed [D][T] so its B-frags are contiguous loads.
// Writes y16 [N*T][H*D] f16 for the output projection.
// ---------------------------------------------------------------------------
__global__ void __launch_bounds__(256)
attn_kernel(const _Float16* __restrict__ q16, const _Float16* __restrict__ k16,
            const _Float16* __restrict__ v16t, _Float16* __restrict__ y16) {
  __shared__ __align__(16) _Float16 Pbuf[8][16 * 32];

  const int batch = blockIdx.y;                 // n*H + h
  const int n = batch / H_N, h = batch % H_N;
  const _Float16* Q  = q16  + (size_t)batch * T_S * D_H;
  const _Float16* Kp = k16  + (size_t)batch * T_S * D_H;
  const _Float16* Vt = v16t + (size_t)batch * (size_t)D_H * T_S;

  const int w    = threadIdx.x >> 5;
  const int lane = threadIdx.x & 31;
  const int col  = lane & 15;
  const int rb   = (lane & 16) ? 8 : 0;
  const int q0   = (blockIdx.x * 8 + w) * 16;
  _Float16* Pw = &Pbuf[w][0];

  v16h qf[4];
#pragma unroll
  for (int kc = 0; kc < 4; ++kc) qf[kc] = load_a_frag(Q, D_H, q0, kc * 32);

  v8f o[8] = {};
  float mstate[8], lstate[8];
#pragma unroll
  for (int r = 0; r < 8; ++r) { mstate[r] = -INFINITY; lstate[r] = 0.0f; }

  const float scale = 0.08838834764831845f;     // 1/sqrt(128)

  for (int s0 = 0; s0 < q0 + 16; s0 += 32) {
    // S = Q * K^T over D=128 (two 16x16 score tiles)
    v8f sc0 = {}, sc1 = {};
#pragma unroll
    for (int kc = 0; kc < 4; ++kc) {
      const v16h b0 = load_b_frag(Kp, D_H, s0,      kc * 32);
      const v16h b1 = load_b_frag(Kp, D_H, s0 + 16, kc * 32);
      sc0 = wmma16(qf[kc], b0, sc0);
      sc1 = wmma16(qf[kc], b1, sc1);
    }
    // online softmax per query row (row lives across a 16-lane half)
#pragma unroll
    for (int r = 0; r < 8; ++r) {
      const int qrow = q0 + rb + r;
      float s0v = (s0 + col      <= qrow) ? sc0[r] * scale : -INFINITY;
      float s1v = (s0 + 16 + col <= qrow) ? sc1[r] * scale : -INFINITY;
      float mx = fmaxf(s0v, s1v);
#pragma unroll
      for (int off = 8; off > 0; off >>= 1) mx = fmaxf(mx, __shfl_xor(mx, off, 32));
      const float mnew  = fmaxf(mstate[r], mx);
      const float alpha = __expf(mstate[r] - mnew);
      const float p0 = __expf(s0v - mnew);
      const float p1 = __expf(s1v - mnew);
      float psum = p0 + p1;
#pragma unroll
      for (int off = 8; off > 0; off >>= 1) psum += __shfl_xor(psum, off, 32);
      lstate[r] = lstate[r] * alpha + psum;
      mstate[r] = mnew;
#pragma unroll
      for (int ni = 0; ni < 8; ++ni) o[ni][r] *= alpha;
      Pw[(rb + r) * 32 + col]      = (_Float16)p0;   // C-layout -> LDS
      Pw[(rb + r) * 32 + 16 + col] = (_Float16)p1;
    }
    // reload P in A-layout (same-wave LDS ops are in-order via DScnt)
    const v16h pa = load_a_frag(Pw, 32, 0, 0);
#pragma unroll
    for (int ni = 0; ni < 8; ++ni) {
      const v16h bv = load_b_frag(Vt, T_S, ni * 16, s0);
      o[ni] = wmma16(pa, bv, o[ni]);
    }
  }

  // epilogue: divide by row sums, write y16[n*T + t][h*D + d]
#pragma unroll
  for (int r = 0; r < 8; ++r) {
    const float inv = 1.0f / lstate[r];
    const size_t t = (size_t)(q0 + rb + r);
    _Float16* yr = y16 + ((size_t)n * T_S + t) * (H_N * D_H) + (size_t)h * D_H;
#pragma unroll
    for (int ni = 0; ni < 8; ++ni)
      yr[ni * 16 + col] = (_Float16)(o[ni][r] * inv);
  }
}

// ---------------------------------------------------------------------------
extern "C" void kernel_launch(void* const* d_in, const int* in_sizes, int n_in,
                              void* d_out, int out_size, void* d_ws, size_t ws_size,
                              hipStream_t stream) {
  const float* x  = (const float*)d_in[0];
  const float* wq = (const float*)d_in[1];
  const float* wk = (const float*)d_in[2];
  const float* wv = (const float*)d_in[3];
  const float* wo = (const float*)d_in[4];
  float* out = (float*)d_out;

  const size_t NTC  = (size_t)N_B * T_S * C_D;            // 8,388,608
  const size_t HDC  = (size_t)H_N * D_H * C_D;            // 4,194,304
  const size_t NHTD = (size_t)N_B * H_N * T_S * D_H;      // 8,388,608

  // workspace layout (f16 halves); total = 2*(NTC + 4*HDC + 3*NHTD) = 96 MiB
  _Float16* x16   = (_Float16*)d_ws;
  _Float16* wq16  = x16   + NTC;
  _Float16* wk16  = wq16  + HDC;
  _Float16* wv16  = wk16  + HDC;
  _Float16* wo16t = wv16  + HDC;        // transposed: [C][H*D]
  _Float16* q16   = wo16t + HDC;        // [batch=n*H+h][T][D]
  _Float16* k16   = q16   + NHTD;       // [batch][T][D]
  _Float16* v16t  = k16   + NHTD;       // [batch][D][T] (transposed)
  _Float16* y16   = x16;                // x16 dead after projections -> reuse

  float* k_out = out + NTC;             // outputs: (out, k, v) concatenated
  float* v_out = out + 2 * NTC;

  const dim3 blk(256);

  // 1) one-time fp32 -> f16 staging (L2-resident thereafter)
  cvt_f16_kernel<<<512, 256, 0, stream>>>(x,  x16,  NTC);
  cvt_f16_kernel<<<256, 256, 0, stream>>>(wq, wq16, HDC);
  cvt_f16_kernel<<<256, 256, 0, stream>>>(wk, wk16, HDC);
  cvt_f16_kernel<<<256, 256, 0, stream>>>(wv, wv16, HDC);
  cvt_f16_t_kernel<<<256, 256, 0, stream>>>(wo, wo16t, H_N * D_H, C_D);

  // 2) QKV projections: per (n,h) batch, M=T, N=D, K=C
  {
    const dim3 grid((T_S / 32) * (D_H / 64) / 8, N_B * H_N);
    // Q: f16 only
    wmma_gemm_kernel<<<grid, blk, 0, stream>>>(
        x16, wq16, q16, nullptr, T_S, D_H, C_D, C_D, C_D,
        (long)T_S * C_D, (long)D_H * C_D, H_N,
        (long)T_S * D_H, D_H, 0, 0L, 0);
    // K: f16 [T][D] + fp32 output k
    wmma_gemm_kernel<<<grid, blk, 0, stream>>>(
        x16, wk16, k16, k_out, T_S, D_H, C_D, C_D, C_D,
        (long)T_S * C_D, (long)D_H * C_D, H_N,
        (long)T_S * D_H, D_H, 0, (long)T_S * D_H, D_H);
    // V: f16 transposed [D][T] + fp32 output v
    wmma_gemm_kernel<<<grid, blk, 0, stream>>>(
        x16, wv16, v16t, v_out, T_S, D_H, C_D, C_D, C_D,
        (long)T_S * C_D, (long)D_H * C_D, H_N,
        (long)D_H * T_S, T_S, 1, (long)T_S * D_H, D_H);
  }

  // 3) causal flash attention -> y16 [N*T][H*D]
  {
    const dim3 grid(T_S / 128, N_B * H_N);
    attn_kernel<<<grid, blk, 0, stream>>>(q16, k16, v16t, y16);
  }

  // 4) output projection: out = Y [N*T, H*D] x Wo^T-rows, M=4096 N=C K=2048
  {
    const dim3 grid(((N_B * T_S) / 32) * (C_D / 64) / 8, 1);
    wmma_gemm_kernel<<<grid, blk, 0, stream>>>(
        y16, wo16t, nullptr, out, N_B * T_S, C_D, H_N * D_H,
        H_N * D_H, H_N * D_H,
        0L, 0L, 1,
        0L, 0, 0, 0L, C_D);
  }
}
